// BiasedAttention_31301721653690
// MI455X (gfx1250) — matrix operand order repaired and checked by reading, not compiled, not run on hardware
//
#include <hip/hip_runtime.h>

typedef __attribute__((ext_vector_type(16))) _Float16 v16h;
typedef __attribute__((ext_vector_type(2)))  _Float16 h2v;
typedef __attribute__((ext_vector_type(8)))  float    v8f;

#define B_  2
#define H_  16
#define S_  2048
#define D_  64
#define QT  16                 // q rows per wave
#define WAVES 8                // waves per block
#define QROWS (QT*WAVES)       // 128 q rows per block
#define JT  32                 // keys per iteration
#define NT  (S_/JT)            // 64 key tiles
#define K_STRIDE  68           // f16 units, padded for bank-conflict-free reads
#define VT_STRIDE 36
#define P_STRIDE  34
#define BIAS_ROWF 36           // floats per bias row (32 + 4 pad -> 144B, 16B aligned)
#define BIAS_TILEF (16*BIAS_ROWF)       // floats per bias tile (one wave)
#define NEG_MAX 3.402823466e38f
#define LOG2E 1.4426950408889634f

__global__ __launch_bounds__(256)
void fa_fwd_kernel(const float* __restrict__ qg, const float* __restrict__ kg,
                   const float* __restrict__ vg, const unsigned char* __restrict__ maskg,
                   const float* __restrict__ tausg, const float* __restrict__ biasg,
                   float* __restrict__ outg)
{
    __shared__ _Float16 lds_k [JT * K_STRIDE];   // K tile, row-major [j][d]
    __shared__ _Float16 lds_vt[D_ * VT_STRIDE];  // V tile, transposed [d][j]
    __shared__ _Float16 lds_p [WAVES * QT * P_STRIDE]; // per-wave P staging [m][j]
    __shared__ __align__(16) float lds_bias[2 * WAVES * BIAS_TILEF]; // double-buffered bias tiles

    const int tid  = threadIdx.x;
    const int lane = tid & 31;
    const int wave = tid >> 5;
    const int half = lane >> 4;   // 0 or 1
    const int ln   = lane & 15;

    const int blk  = blockIdx.x;            // 0..511
    const int qblk = blk % (S_ / QROWS);    // 16 q-blocks
    const int bh   = blk / (S_ / QROWS);    // 0..31
    const int b    = bh / H_;
    const int h    = bh % H_;
    const int i0   = qblk * QROWS + wave * QT;

    // fold log2(e) into tau and the 1/sqrt(D) Q scale: all scores live in log2 domain
    const float taul = tausg[h] * LOG2E;
    const float* qp = qg + (((size_t)bh) * S_ + i0) * D_;
    const float* kp = kg + ((size_t)bh) * S_ * D_;
    const float* vp = vg + ((size_t)bh) * S_ * D_;
    const unsigned char* maskp = maskg + (size_t)b * S_;
    const float* biasp = biasg + (size_t)b * S_ * S_;

    // ---- async bias double-buffer setup ------------------------------------
    // One tile = 16 rows x 32 cols fp32 for this wave's q rows. 4 async B128 ops
    // per tile: op i covers rows 4i..4i+3; lane covers row (lane>>3), 16B chunk (lane&7).
    const unsigned lane_row  = lane >> 3;        // 0..3 within a 4-row slab
    const unsigned lane_col  = (lane & 7) * 16;  // byte offset within row
    const unsigned lds_bias_base = (unsigned)(size_t)(&lds_bias[0]);
    const unsigned ptile0 = lds_bias_base + (unsigned)wave * (BIAS_TILEF * 4)
                          + lane_row * (BIAS_ROWF * 4) + lane_col;      // buffer 0
    const unsigned ptile1 = ptile0 + (unsigned)(WAVES * BIAS_TILEF * 4); // buffer 1
    const unsigned long long sbias = (unsigned long long)(size_t)biasp;
    unsigned vbase = (unsigned)((unsigned)(i0 + (int)lane_row) * (S_ * 4)) + lane_col;

    // prologue: async-load bias tile 0 into buffer 0
    #pragma unroll
    for (int i = 0; i < 4; ++i) {
        asm volatile("global_load_async_to_lds_b128 %0, %1, %2"
                     :: "v"(ptile0 + (unsigned)(i * BIAS_ROWF * 16)),
                        "v"(vbase + (unsigned)(i * 4 * S_ * 4)),
                        "s"(sbias) : "memory");
    }
    vbase += JT * 4;

    // ---- load Q A-fragments (16x32 halves of D=64), log2e/sqrt(D) folded in ----
    // A layout: lane holds row m = ln; K packed {0..7,16..23} (half=0) / {8..15,24..31} (half=1)
    v16h a0, a1;
    {
        const float qscale = 0.125f * LOG2E;
        const float* qrow = qp + (size_t)ln * D_;
        #pragma unroll
        for (int p = 0; p < 8; ++p) {
            const int kk = 2*p + (p >= 4 ? 8 : 0) + half*8;
            float2 t0 = *(const float2*)(qrow + kk);
            float2 t1 = *(const float2*)(qrow + kk + 32);
            a0[2*p]   = (_Float16)(t0.x * qscale);
            a0[2*p+1] = (_Float16)(t0.y * qscale);
            a1[2*p]   = (_Float16)(t1.x * qscale);
            a1[2*p+1] = (_Float16)(t1.y * qscale);
        }
    }

    // constant all-ones B fragment: P(16x32) x ones(32x16) = row sums in C layout
    v16h ones;
    #pragma unroll
    for (int e = 0; e < 16; ++e) ones[e] = (_Float16)1.0f;

    v8f acc[4] = {};               // O accumulator, 16x64 fp32 (4 N-chunks)
    v8f lacc   = {};               // row-sum accumulator (replicated over lanes)
    float mrun = -NEG_MAX;         // running tile max (log2 domain), wave-uniform

    _Float16* prow = &lds_p[wave * QT * P_STRIDE];

    // K/V staging: 256 threads cover 32x64 tile as 1024 float2; 4 per thread.
    // pair index e = rr*256 + tid -> row = e>>5, float2-in-row = e&31
    const int st_row = tid >> 5;          // base row for rr=0 (rows step by 8 per rr)
    const int st_c2  = tid & 31;          // float2 index within row
    const float* kjt = kp + (size_t)st_row * D_ + st_c2 * 2;
    const float* vjt = vp + (size_t)st_row * D_ + st_c2 * 2;

    for (int jt = 0; jt < NT; ++jt) {
        const int j0 = jt * JT;

        // ---- kick off async bias load for the NEXT tile (double buffer) ----
        if (jt + 1 < NT) {
            const unsigned pt = ((jt + 1) & 1) ? ptile1 : ptile0;
            #pragma unroll
            for (int i = 0; i < 4; ++i) {
                asm volatile("global_load_async_to_lds_b128 %0, %1, %2"
                             :: "v"(pt + (unsigned)(i * BIAS_ROWF * 16)),
                                "v"(vbase + (unsigned)(i * 4 * S_ * 4)),
                                "s"(sbias) : "memory");
            }
            vbase += JT * 4;
        }

        __syncthreads();   // previous iteration's LDS reads done
        // ---- cooperative stage of K (row-major f16) and V (transposed f16) ----
        // packed f32x2 loads + v_cvt_pk_rtz_f16_f32, 4 pairs each for K and V
        #pragma unroll
        for (int rr = 0; rr < 4; ++rr) {
            const int row = st_row + rr * 8;          // e>>5 with e = rr*256+tid
            const int col = st_c2 * 2;
            float2 kv2 = *(const float2*)(kjt + (size_t)rr * 8 * D_);
            float2 vv2 = *(const float2*)(vjt + (size_t)rr * 8 * D_);
            h2v pk = __builtin_bit_cast(h2v, __builtin_amdgcn_cvt_pkrtz(kv2.x, kv2.y));
            h2v pv = __builtin_bit_cast(h2v, __builtin_amdgcn_cvt_pkrtz(vv2.x, vv2.y));
            *(h2v*)(&lds_k[row * K_STRIDE + col]) = pk;          // b32 store
            lds_vt[(col    ) * VT_STRIDE + row] = pv.x;          // b16 store
            lds_vt[(col + 1) * VT_STRIDE + row] = pv.y;          // b16_d16_hi store
        }
        kjt += JT * D_;
        vjt += JT * D_;
        if (j0 + JT < S_) { // prefetch next K/V tile into caches
            __builtin_prefetch(kjt + tid * 8, 0, 0);
            __builtin_prefetch(vjt + tid * 8, 0, 0);
        }
        __syncthreads();

        // ---- S = log2e * (Q K^T / sqrt(D)) : two 16x16 chunks via 4 WMMAs ----
        v8f sc[2];
        #pragma unroll
        for (int c = 0; c < 2; ++c) {
            const _Float16* krow = &lds_k[(c*16 + ln) * K_STRIDE];
            v16h b0, b1;
            #pragma unroll
            for (int p = 0; p < 8; ++p) {
                const int kk = 2*p + (p >= 4 ? 8 : 0) + half*8;   // B layout: K packed {0..7,16..23} (half=0) / {8..15,24..31} (half=1)
                h2v t0 = *(const h2v*)(krow + kk);
                h2v t1 = *(const h2v*)(krow + kk + 32);
                b0[2*p] = t0.x; b0[2*p+1] = t0.y;
                b1[2*p] = t1.x; b1[2*p+1] = t1.y;
            }
            v8f z = {};
            v8f s0 = __builtin_amdgcn_wmma_f32_16x16x32_f16(false, a0, false, b0,
                                                            (short)0, z, false, false);
            sc[c]  = __builtin_amdgcn_wmma_f32_16x16x32_f16(false, a1, false, b1,
                                                            (short)0, s0, false, false);
        }

        // ---- wait for THIS tile's async bias (prev 4 ops); next tile's 4 stay in flight
        if (jt + 1 < NT) {
            asm volatile("s_wait_asynccnt 0x4" ::: "memory");
        } else {
            asm volatile("s_wait_asynccnt 0x0" ::: "memory");
        }
        const float* btile = lds_bias + ((jt & 1) ? WAVES * BIAS_TILEF : 0)
                                      + wave * BIAS_TILEF;

        // ---- mask + tau*bias, C-layout: element r -> row m=r+8*half, col = ln+16*c ----
        #pragma unroll
        for (int c = 0; c < 2; ++c) {
            const int j = j0 + c*16 + ln;
            const bool keep = maskp[j] != 0;
            #pragma unroll
            for (int r = 0; r < 8; ++r) {
                const int m = r + 8*half;
                const float bv = btile[m * BIAS_ROWF + ln + 16*c];
                float s = sc[c][r];
                s = keep ? s : -NEG_MAX;
                sc[c][r] = s - taul * bv;
            }
        }

        // ---- online softmax, tile-scoped max (softmax is invariant to the shift) ----
        float tmax = fmaxf(sc[0][0], sc[1][0]);
        #pragma unroll
        for (int r = 1; r < 8; ++r) tmax = fmaxf(tmax, fmaxf(sc[0][r], sc[1][r]));
        #pragma unroll
        for (int off = 1; off < 32; off <<= 1)
            tmax = fmaxf(tmax, __shfl_xor(tmax, off, 32));   // wave-wide: 5 cross-lane ops

        const float mold = mrun;
        mrun = fmaxf(mrun, tmax);
        const float corr = __builtin_amdgcn_exp2f(mold - mrun);
        // corr is wave-uniform: skip the rescale scalar-side when max didn't grow
        if (__builtin_amdgcn_readfirstlane(__float_as_uint(corr)) != 0x3f800000u) {
            #pragma unroll
            for (int c2 = 0; c2 < 4; ++c2)
                #pragma unroll
                for (int r = 0; r < 8; ++r)
                    acc[c2][r] *= corr;
            #pragma unroll
            for (int r = 0; r < 8; ++r) lacc[r] *= corr;
        }

        // ---- P = exp2(S - m), stage P to LDS in [m][j] order ----
        #pragma unroll
        for (int c = 0; c < 2; ++c) {
            #pragma unroll
            for (int r = 0; r < 8; ++r) {
                const int m = r + 8*half;
                const int n = ln + 16*c;
                const float pv = __builtin_amdgcn_exp2f(sc[c][r] - mrun);
                prow[m * P_STRIDE + n] = (_Float16)pv;
            }
        }

        // in-wave LDS ordering: P stores must land before cross-lane reads
        asm volatile("s_wait_dscnt 0x0" ::: "memory");

        // ---- read P back as A-fragment (transpose via LDS) ----
        v16h pa;
        {
            const _Float16* pr = prow + ln * P_STRIDE;
            #pragma unroll
            for (int p = 0; p < 8; ++p) {
                const int kk = 2*p + (p >= 4 ? 8 : 0) + half*8;
                h2v t = *(const h2v*)(pr + kk);
                pa[2*p] = t.x; pa[2*p+1] = t.y;
            }
        }

        // ---- row sums on the matrix pipe: lacc += P x ones ----
        lacc = __builtin_amdgcn_wmma_f32_16x16x32_f16(false, pa, false, ones,
                                                      (short)0, lacc, false, false);

        // ---- O += P V : 4 WMMAs over the 4 d-chunks ----
        #pragma unroll
        for (int c2 = 0; c2 < 4; ++c2) {
            const _Float16* vtrow = &lds_vt[(c2*16 + ln) * VT_STRIDE];
            v16h vb;
            #pragma unroll
            for (int p = 0; p < 8; ++p) {
                const int kk = 2*p + (p >= 4 ? 8 : 0) + half*8;
                h2v t = *(const h2v*)(vtrow + kk);
                vb[2*p] = t.x; vb[2*p+1] = t.y;
            }
            acc[c2] = __builtin_amdgcn_wmma_f32_16x16x32_f16(false, pa, false, vb,
                                                             (short)0, acc[c2], false, false);
        }
    }

    // ---- epilogue: O / rowsum, fp32 store ----
    float rl[8];
    #pragma unroll
    for (int r = 0; r < 8; ++r) rl[r] = 1.0f / lacc[r];
    float* op = outg + ((size_t)bh * S_ + i0) * D_;
    #pragma unroll
    for (int c2 = 0; c2 < 4; ++c2) {
        #pragma unroll
        for (int r = 0; r < 8; ++r) {
            const int m = r + 8*half;
            op[(size_t)m * D_ + c2*16 + ln] = acc[c2][r] * rl[r];
        }
    }
}

extern "C" void kernel_launch(void* const* d_in, const int* in_sizes, int n_in,
                              void* d_out, int out_size, void* d_ws, size_t ws_size,
                              hipStream_t stream) {
    const float* q            = (const float*)d_in[0];
    const float* k            = (const float*)d_in[1];
    const float* v            = (const float*)d_in[2];
    const unsigned char* mask = (const unsigned char*)d_in[3]; // jnp bool = 1 byte
    const float* taus         = (const float*)d_in[4];
    const float* bias         = (const float*)d_in[5];
    float* out                = (float*)d_out;

    dim3 grid(B_ * H_ * (S_ / QROWS));   // 512 blocks
    dim3 block(256);                     // 8 waves of 32
    fa_fwd_kernel<<<grid, block, 0, stream>>>(q, k, v, mask, taus, bias, out);
}
